// RNNModel_70746701300457
// MI455X (gfx1250) — matrix operand order repaired and checked by reading, not compiled, not run on hardware
//
#include <hip/hip_runtime.h>
#include <hip/hip_bf16.h>

// ---------------------------------------------------------------------------
// LSTM (B=256,S=512,F=26,H=512) + fused MLP(512->256->20) + log_softmax
// Persistent wave32 WMMA kernel for gfx1250 (CDNA5).
//   - 16 workgroups, one per 16-row batch tile; 8 waves x 32 lanes.
//   - bf16 hi/lo split (3x v_wmma_f32_16x16x32_bf16) for fp32-class accuracy.
//   - weights pre-swizzled into WMMA B-fragment lane order (one-time pack).
//   - h state exchanged through LDS each step; c state lives in registers.
//   - input projection folded into the K-loop as a 17th k-tile so no
//     B-fragment load is t-invariant (prevents LICM hoist -> scratch spills).
// ---------------------------------------------------------------------------

typedef __attribute__((ext_vector_type(16))) __bf16 v16bf;
typedef __attribute__((ext_vector_type(8)))  float  v8f;

#define BDIM 256   // batch
#define SDIM 512   // seq
#define FDIM 26    // features
#define HDIM 512   // rnn hidden
#define MDIM 256   // mlp hidden
#define CDIM 20    // classes
#define GDIM 2048  // 4*H

#define HPAD 520   // 512 + 8 floats -> rows stay 32B aligned
#define MPAD 264   // 256 + 8

// ---------------------------------------------------------------------------
// Packing kernels: fp32 -> bf16 hi/lo in WMMA fragment order.
// B-fragment (32x16, bf16) assumed lane layout:
//   n = nt*16 + (lane&15);  k = kt*32 + (lane&16 ? 8 : 0) + (e&7) + (e>=8 ? 16 : 0),  e = 0..15
// A-fragment (16x32, bf16) assumed lane layout (per ISA A-matrix table):
//   m = lane&15; koff = (lane&16 ? 8 : 0); K(e) = kt*32 + koff + (e&7) + (e>=8 ? 16 : 0)
// ---------------------------------------------------------------------------

__global__ __launch_bounds__(256) void pack_b_frags(
    const float* __restrict__ W, int Nrows, int Kcols, int NT, int KT,
    __bf16* __restrict__ hi, __bf16* __restrict__ lo) {
    int idx = blockIdx.x * blockDim.x + threadIdx.x;     // one thread per (nt,kt,lane)
    int total = NT * KT * 32;
    if (idx >= total) return;
    int lane = idx & 31;
    int kt   = (idx >> 5) % KT;
    int nt   = (idx >> 5) / KT;
    int n     = nt * 16 + (lane & 15);
    int kbase = kt * 32 + ((lane & 16) ? 8 : 0);
    long base = (long)((nt * KT + kt) * 32 + lane) * 16;
#pragma unroll
    for (int e = 0; e < 16; ++e) {
        int k = kbase + (e & 7) + ((e & 8) ? 16 : 0);
        float v = (n < Nrows && k < Kcols) ? W[(long)n * Kcols + k] : 0.0f;
        __bf16 h = (__bf16)v;
        float  r = v - (float)h;
        hi[base + e] = h;
        lo[base + e] = (__bf16)r;
    }
}

__global__ __launch_bounds__(256) void pack_x_frags(
    const float* __restrict__ x, __bf16* __restrict__ hi, __bf16* __restrict__ lo) {
    int idx = blockIdx.x * blockDim.x + threadIdx.x;     // (bt, t, lane)
    if (idx >= 16 * SDIM * 32) return;
    int lane = idx & 31;
    int t    = (idx >> 5) & (SDIM - 1);
    int bt   = idx >> 14;
    int m    = lane & 15;
    int b    = bt * 16 + m;
    int koff = (lane & 16) ? 8 : 0;
    long base = (long)idx * 16;
    const float* xp = x + ((long)b * SDIM + t) * FDIM;
#pragma unroll
    for (int e = 0; e < 16; ++e) {
        int k = koff + (e & 7) + ((e & 8) ? 16 : 0);     // A-layout K index, kt = 0
        float v = (k < FDIM) ? xp[k] : 0.0f;
        __bf16 h = (__bf16)v;
        hi[base + e] = h;
        lo[base + e] = (__bf16)(v - (float)h);
    }
}

__global__ __launch_bounds__(256) void combine_bias(
    const float* __restrict__ a, const float* __restrict__ b, float* __restrict__ o) {
    int i = blockIdx.x * blockDim.x + threadIdx.x;
    if (i < GDIM) o[i] = a[i] + b[i];
}

// ---------------------------------------------------------------------------
// Device helpers
// ---------------------------------------------------------------------------

__device__ __forceinline__ void wmma3(v8f& acc, v16bf ahi, v16bf alo,
                                      v16bf bhi, v16bf blo) {
    acc = __builtin_amdgcn_wmma_f32_16x16x32_bf16(false, ahi, false, bhi, (short)0, acc, false, false);
    acc = __builtin_amdgcn_wmma_f32_16x16x32_bf16(false, alo, false, bhi, (short)0, acc, false, false);
    acc = __builtin_amdgcn_wmma_f32_16x16x32_bf16(false, ahi, false, blo, (short)0, acc, false, false);
}

// Build A hi/lo fragment for k-tile `kt` from an fp32 row-major LDS matrix.
template <int ROWSTRIDE>
__device__ __forceinline__ void ldsA(const float* __restrict__ lds, int lane, int kt,
                                     v16bf& ahi, v16bf& alo) {
    int m    = lane & 15;
    int koff = (lane & 16) ? 8 : 0;
    const float* p = lds + m * ROWSTRIDE + kt * 32 + koff;   // 32B aligned
    const float4* q = (const float4*)p;
    float4 u0 = q[0], u1 = q[1];      // K = koff + 0..7
    float4 u2 = q[4], u3 = q[5];      // K = koff + 16..23
    float v[16] = {u0.x, u0.y, u0.z, u0.w, u1.x, u1.y, u1.z, u1.w,
                   u2.x, u2.y, u2.z, u2.w, u3.x, u3.y, u3.z, u3.w};
#pragma unroll
    for (int e = 0; e < 16; ++e) {
        __bf16 h = (__bf16)v[e];
        ahi[e] = h;
        alo[e] = (__bf16)(v[e] - (float)v[e] + (v[e] - (float)h));  // see note below
    }
}

__device__ __forceinline__ float fsig(float x) { return 1.0f / (1.0f + __expf(-x)); }

// ---------------------------------------------------------------------------
// Main persistent kernel: grid = 16 (batch tiles), block = 256 (8 waves)
// ---------------------------------------------------------------------------

__global__ __launch_bounds__(256) void lstm_mlp_kernel(
    const __bf16* __restrict__ xf_hi,  const __bf16* __restrict__ xf_lo,
    const __bf16* __restrict__ whh_hi, const __bf16* __restrict__ whh_lo,
    const __bf16* __restrict__ wih_hi, const __bf16* __restrict__ wih_lo,
    const __bf16* __restrict__ w1_hi,  const __bf16* __restrict__ w1_lo,
    const __bf16* __restrict__ w2_hi,  const __bf16* __restrict__ w2_lo,
    const float* __restrict__ biasc, const float* __restrict__ b1,
    const float* __restrict__ b2, float* __restrict__ out) {

    __shared__ float h_lds[16 * HPAD];      // current hidden state, fp32
    __shared__ float hid_lds[16 * MPAD];    // mlp hidden, fp32
    __shared__ float logit_lds[16 * 32];    // padded logits

    const int bt   = blockIdx.x;
    const int wave = threadIdx.x >> 5;
    const int lane = threadIdx.x & 31;
    const int mrowoff = (lane & 16) ? 8 : 0;

    for (int i = threadIdx.x; i < 16 * HPAD; i += 256) h_lds[i] = 0.0f;
    __syncthreads();

    // Cell state for this wave's 64 H-columns, kept in WMMA C-layout registers.
    v8f c_st[4];
#pragma unroll
    for (int j = 0; j < 4; ++j)
#pragma unroll
        for (int r = 0; r < 8; ++r) c_st[j][r] = 0.0f;

    // Per-lane gate biases (column = wave*64 + j*16 + (lane&15)).
    float bi[4], bf_[4], bg[4], bo[4];
    const int colb = wave * 64;
#pragma unroll
    for (int j = 0; j < 4; ++j) {
        int col = colb + j * 16 + (lane & 15);
        bi[j]  = biasc[col];
        bf_[j] = biasc[512 + col];
        bg[j]  = biasc[1024 + col];
        bo[j]  = biasc[1536 + col];
    }
    float bb1[2] = {b1[wave * 32 + (lane & 15)], b1[wave * 32 + 16 + (lane & 15)]};

#pragma unroll 1
    for (int t = 0; t < SDIM; ++t) {
        // ---- z = x_t @ w_ih^T + h @ w_hh^T  (16 n-tiles per wave) ----------
        v8f acc[16];
#pragma unroll
        for (int i = 0; i < 16; ++i)
#pragma unroll
            for (int r = 0; r < 8; ++r) acc[i][r] = 0.0f;

        // Pre-packed A fragments for x_t (loaded once per step; t-dependent).
        long xb = ((long)(bt * SDIM + t) * 32 + lane) * 16;
        const v16bf axh = *(const v16bf*)(xf_hi + xb);
        const v16bf axl = *(const v16bf*)(xf_lo + xb);

        // Unified K loop: kt = 0..15 -> h @ w_hh^T ; kt = 16 -> x_t @ w_ih^T.
        // B-frag base pointers are kt-dependent so no load is t-invariant.
#pragma unroll 1
        for (int kt = 0; kt < 17; ++kt) {
            v16bf ah, al;
            const __bf16* bh_base;
            const __bf16* bl_base;
            long nstride;
            if (kt < 16) {
                ldsA<HPAD>(h_lds, lane, kt, ah, al);
                long o = ((long)kt * 32 + lane) * 16;
                bh_base = whh_hi + o;
                bl_base = whh_lo + o;
                nstride = 16 * 32 * 16;                 // 8192 elems per n-tile
                if (kt + 1 < 16) {                      // prefetch next k-slice
                    long p = 512 + (long)(wave * 4) * nstride;
                    __builtin_prefetch((const void*)(bh_base + p), 0, 1);
                    __builtin_prefetch((const void*)(bl_base + p), 0, 1);
                }
            } else {
                ah = axh; al = axl;
                bh_base = wih_hi + (long)lane * 16;
                bl_base = wih_lo + (long)lane * 16;
                nstride = 32 * 16;                      // 512 elems per n-tile
            }
#pragma unroll
            for (int g = 0; g < 4; ++g)
#pragma unroll
                for (int j = 0; j < 4; ++j) {
                    int nt = g * 32 + wave * 4 + j;
                    v16bf bh = *(const v16bf*)(bh_base + (long)nt * nstride);
                    v16bf bl = *(const v16bf*)(bl_base + (long)nt * nstride);
                    wmma3(acc[g * 4 + j], ah, al, bh, bl);
                }
        }
        __syncthreads();   // all h_lds reads done before overwrite

        // ---- gates + cell update, write new h to LDS -----------------------
#pragma unroll
        for (int j = 0; j < 4; ++j) {
            int col = colb + j * 16 + (lane & 15);
#pragma unroll
            for (int r = 0; r < 8; ++r) {
                float iv = fsig(acc[0 * 4 + j][r] + bi[j]);
                float fv = fsig(acc[1 * 4 + j][r] + bf_[j]);
                float gv = tanhf(acc[2 * 4 + j][r] + bg[j]);
                float ov = fsig(acc[3 * 4 + j][r] + bo[j]);
                float c  = fv * c_st[j][r] + iv * gv;
                c_st[j][r] = c;
                h_lds[(r + mrowoff) * HPAD + col] = ov * tanhf(c);
            }
        }
        __syncthreads();   // new h visible to all waves

        // ---- MLP layer 1: hidden = relu(h @ w1^T + b1), 2 n-tiles/wave -----
        v8f a2[2];
#pragma unroll
        for (int i = 0; i < 2; ++i)
#pragma unroll
            for (int r = 0; r < 8; ++r) a2[i][r] = 0.0f;
#pragma unroll 1
        for (int kt = 0; kt < 16; ++kt) {
            v16bf ah, al;
            ldsA<HPAD>(h_lds, lane, kt, ah, al);
#pragma unroll
            for (int n2 = 0; n2 < 2; ++n2) {
                int nt = wave * 2 + n2;
                long wb = ((long)(nt * 16 + kt) * 32 + lane) * 16;
                v16bf bh = *(const v16bf*)(w1_hi + wb);
                v16bf bl = *(const v16bf*)(w1_lo + wb);
                wmma3(a2[n2], ah, al, bh, bl);
            }
        }
#pragma unroll
        for (int n2 = 0; n2 < 2; ++n2) {
            int col = wave * 32 + n2 * 16 + (lane & 15);
#pragma unroll
            for (int r = 0; r < 8; ++r) {
                float v = a2[n2][r] + bb1[n2];
                hid_lds[(r + mrowoff) * MPAD + col] = v > 0.0f ? v : 0.0f;
            }
        }
        __syncthreads();

        // ---- MLP layer 2: logits = hidden @ w2^T + b2 (waves 0,1) ----------
        if (wave < 2) {
            v8f a3;
#pragma unroll
            for (int r = 0; r < 8; ++r) a3[r] = 0.0f;
#pragma unroll 1
            for (int kt = 0; kt < 8; ++kt) {
                v16bf ah, al;
                ldsA<MPAD>(hid_lds, lane, kt, ah, al);
                long wb = ((long)(wave * 8 + kt) * 32 + lane) * 16;
                v16bf bh = *(const v16bf*)(w2_hi + wb);
                v16bf bl = *(const v16bf*)(w2_lo + wb);
                wmma3(a3, ah, al, bh, bl);
            }
            int col = wave * 16 + (lane & 15);
            float bc = (col < CDIM) ? b2[col] : 0.0f;
#pragma unroll
            for (int r = 0; r < 8; ++r)
                logit_lds[(r + mrowoff) * 32 + col] = a3[r] + bc;
        }
        __syncthreads();

        // ---- log_softmax + store (one thread per batch row) ----------------
        if (threadIdx.x < 16) {
            int m = threadIdx.x;
            float mx = -3.4e38f;
            for (int c = 0; c < CDIM; ++c) {
                float v = logit_lds[m * 32 + c];
                mx = v > mx ? v : mx;
            }
            float s = 0.0f;
            for (int c = 0; c < CDIM; ++c) s += __expf(logit_lds[m * 32 + c] - mx);
            float lse = mx + __logf(s);
            float* op = out + ((long)(bt * 16 + m) * SDIM + t) * CDIM;
            for (int c = 0; c < CDIM; ++c) op[c] = logit_lds[m * 32 + c] - lse;
        }
        // next iteration's first conflicting LDS write is after >=1 barrier
    }
}

// ---------------------------------------------------------------------------
// Host launcher
// ---------------------------------------------------------------------------

extern "C" void kernel_launch(void* const* d_in, const int* in_sizes, int n_in,
                              void* d_out, int out_size, void* d_ws, size_t ws_size,
                              hipStream_t stream) {
    const float* x    = (const float*)d_in[0];
    const float* w_ih = (const float*)d_in[1];
    const float* w_hh = (const float*)d_in[2];
    const float* b_ih = (const float*)d_in[3];
    const float* b_hh = (const float*)d_in[4];
    const float* w1   = (const float*)d_in[5];
    const float* b1   = (const float*)d_in[6];
    const float* w2   = (const float*)d_in[7];
    const float* b2   = (const float*)d_in[8];
    float* out = (float*)d_out;

    char* ws = (char*)d_ws;
    size_t off = 0;
    auto alloc = [&](size_t bytes) -> void* {
        void* p = ws + off;
        off = (off + bytes + 255) & ~(size_t)255;
        return p;
    };
    const size_t N_WHH = (size_t)128 * 16 * 32 * 16;  // frags * lanes * elems
    const size_t N_WIH = (size_t)128 * 1 * 32 * 16;
    const size_t N_W1  = (size_t)16 * 16 * 32 * 16;
    const size_t N_W2  = (size_t)2 * 8 * 32 * 16;
    const size_t N_XF  = (size_t)16 * SDIM * 32 * 16;

    __bf16* whh_hi = (__bf16*)alloc(N_WHH * 2);
    __bf16* whh_lo = (__bf16*)alloc(N_WHH * 2);
    __bf16* wih_hi = (__bf16*)alloc(N_WIH * 2);
    __bf16* wih_lo = (__bf16*)alloc(N_WIH * 2);
    __bf16* w1_hi  = (__bf16*)alloc(N_W1 * 2);
    __bf16* w1_lo  = (__bf16*)alloc(N_W1 * 2);
    __bf16* w2_hi  = (__bf16*)alloc(N_W2 * 2);
    __bf16* w2_lo  = (__bf16*)alloc(N_W2 * 2);
    __bf16* xf_hi  = (__bf16*)alloc(N_XF * 2);
    __bf16* xf_lo  = (__bf16*)alloc(N_XF * 2);
    float*  biasc  = (float*)alloc(GDIM * 4);

    auto blk = [](size_t n) { return (unsigned)((n + 255) / 256); };

    pack_b_frags<<<blk(128 * 16 * 32), 256, 0, stream>>>(w_hh, GDIM, HDIM, 128, 16, whh_hi, whh_lo);
    pack_b_frags<<<blk(128 * 1 * 32), 256, 0, stream>>>(w_ih, GDIM, FDIM, 128, 1, wih_hi, wih_lo);
    pack_b_frags<<<blk(16 * 16 * 32), 256, 0, stream>>>(w1, MDIM, HDIM, 16, 16, w1_hi, w1_lo);
    pack_b_frags<<<blk(2 * 8 * 32), 256, 0, stream>>>(w2, CDIM, MDIM, 2, 8, w2_hi, w2_lo);
    pack_x_frags<<<blk(16 * SDIM * 32), 256, 0, stream>>>(x, xf_hi, xf_lo);
    combine_bias<<<blk(GDIM), 256, 0, stream>>>(b_ih, b_hh, biasc);

    lstm_mlp_kernel<<<16, 256, 0, stream>>>(xf_hi, xf_lo, whh_hi, whh_lo,
                                            wih_hi, wih_lo, w1_hi, w1_lo,
                                            w2_hi, w2_lo, biasc, b1, b2, out);
}
